// Pain_46188078301610
// MI455X (gfx1250) — matrix operand order repaired and checked by reading, not compiled, not run on hardware
//
#include <hip/hip_runtime.h>
#include <hip/hip_bf16.h>
#include <math.h>

// ---------------------------------------------------------------------------
// PaiNN forward for MI455X (gfx1250, wave32, WMMA).
//   - dense GEMMs: v_wmma_f32_16x16x32_f16, 64x128 C-tile per 128-thread block,
//     B panel staged in LDS (f16, fragment-major), K-loop fully unrolled
//   - rbf filter GEMM (K=20) fused into the edge kernel via LDS weights
//   - segment sums via global fp32 atomics
// ---------------------------------------------------------------------------

#define N_ATOMS 8192
#define N_EDGES 262144
#define N_MOL   128
#define FDIM    128
#define F3      384
#define RDIM    20
#define NLAYER  3
#define CUTOFF  5.0f
#define PI_F    3.14159265358979323846f

typedef __attribute__((ext_vector_type(16))) _Float16 v16h;
typedef __attribute__((ext_vector_type(8)))  float    v8f;

// ---------------------------------------------------------------------------
// WMMA GEMM: C[M,N] = act(A[M,K] @ W[K,N] + bias[N]),  lda == K, ldw == ldc == N.
// Block = 128 threads = 4 waves. Block tile: 64(M) x 128(N).
//   wave w -> rows [by*64 + w*16, +16);  8 accumulators cover all 128 columns.
// Per K-step (32): stage W panel (32x128) into LDS as f16 in fragment-major
// [col][k] order; A fragment loaded as 4x float4 and converted.
//
// Fragment layouts used here (wave32; contiguous k per lane half, not the ISA table order):
//  A 16x32 f16 : lane&15 = row; lanes>=16 take K=16..31;
//                elem e -> K = koffA + e (16 contiguous)
//  B 32x16 f16 : lane&15 = col; lanes>=16 take K=16..31  -> 16 contiguous
//                f16 at lB[col*32 + hi*16]
//  C/D 16x16 f32: lane&15 = col; VGPR r -> row = r + hi*8
// ---------------------------------------------------------------------------
template <int K, bool SILU, bool BIAS>
__global__ __launch_bounds__(128)
void k_gemm_wmma(const float* __restrict__ A,
                 const float* __restrict__ W,
                 const float* __restrict__ bias,
                 float* __restrict__ C, int N) {
  __shared__ _Float16 lB[128 * 32];   // [col][k], 8 KB

  const int tid   = threadIdx.x;
  const int lane  = tid & 31;
  const int wave  = tid >> 5;
  const int hi    = lane >> 4;        // 0: lanes 0-15, 1: lanes 16-31
  const int l15   = lane & 15;
  const int nb    = blockIdx.x * 128;
  const int m0    = blockIdx.y * 64 + wave * 16;
  const int arow  = m0 + l15;
  const int koffA = hi << 4;

  v8f acc[8] = {};

#pragma unroll
  for (int k0 = 0; k0 < K; k0 += 32) {
    __syncthreads();   // previous iteration's fragment reads done
    // ---- stage W[k0..k0+31, nb..nb+127] -> lB, coalesced global reads ----
#pragma unroll
    for (int j = 0; j < 32; ++j) {
      const int idx = tid + 128 * j;      // 0..4095
      const int kr  = idx >> 7;           // 0..31
      const int nc  = idx & 127;          // 0..127
      lB[nc * 32 + kr] = (_Float16)W[(size_t)(k0 + kr) * N + (nb + nc)];
    }
    __syncthreads();

    // ---- A fragment: 16 f16 per lane via 4x float4 ----
    const float4* ap = (const float4*)(A + (size_t)arow * K + k0 + koffA);
    const float4 a0 = ap[0], a1 = ap[1], a2 = ap[2], a3 = ap[3];
    v16h af;
    af[0]  = (_Float16)a0.x; af[1]  = (_Float16)a0.y;
    af[2]  = (_Float16)a0.z; af[3]  = (_Float16)a0.w;
    af[4]  = (_Float16)a1.x; af[5]  = (_Float16)a1.y;
    af[6]  = (_Float16)a1.z; af[7]  = (_Float16)a1.w;
    af[8]  = (_Float16)a2.x; af[9]  = (_Float16)a2.y;
    af[10] = (_Float16)a2.z; af[11] = (_Float16)a2.w;
    af[12] = (_Float16)a3.x; af[13] = (_Float16)a3.y;
    af[14] = (_Float16)a3.z; af[15] = (_Float16)a3.w;

    // ---- 8 independent WMMA chains across the 128 block columns ----
#pragma unroll
    for (int nt = 0; nt < 8; ++nt) {
      const _Float16* bp = &lB[(nt * 16 + l15) * 32 + hi * 16];
      v16h bf;
#pragma unroll
      for (int e = 0; e < 16; ++e) bf[e] = bp[e];   // 2x 16B contiguous LDS
      acc[nt] = __builtin_amdgcn_wmma_f32_16x16x32_f16(
          false, af, false, bf, (short)0, acc[nt], false, false);
    }
  }

  // ---- epilogue ----
  const int mbase = m0 + (hi << 3);
#pragma unroll
  for (int nt = 0; nt < 8; ++nt) {
    const int col  = nb + nt * 16 + l15;
    const float bv = BIAS ? bias[col] : 0.0f;
#pragma unroll
    for (int r = 0; r < 8; ++r) {
      float x = acc[nt][r] + bv;
      if (SILU) x = x * __builtin_amdgcn_rcpf(1.0f + __expf(-x));  // fast SiLU
      C[(size_t)(mbase + r) * N + col] = x;
    }
  }
}

// ---------------------------------------------------------------------------
// s = embed[z], v = 0
// ---------------------------------------------------------------------------
__global__ void k_embed(const int* __restrict__ z, const float* __restrict__ embed,
                        float* __restrict__ s, float* __restrict__ v) {
  const int i = blockIdx.x, f = threadIdx.x;
  s[(size_t)i * FDIM + f] = embed[(size_t)z[i] * FDIM + f];
  v[((size_t)i * 3 + 0) * FDIM + f] = 0.0f;
  v[((size_t)i * 3 + 1) * FDIM + f] = 0.0f;
  v[((size_t)i * 3 + 2) * FDIM + f] = 0.0f;
}

// ---------------------------------------------------------------------------
// Per-edge geometry: unit vector, cosine cutoff, 20 sinc RBFs (reused x3 layers)
// ---------------------------------------------------------------------------
__global__ void k_edge_geom(const float* __restrict__ pos,
                            const int* __restrict__ esrc, const int* __restrict__ edst,
                            float* __restrict__ unitv, float* __restrict__ fcv,
                            float* __restrict__ rbfv) {
  const int e = blockIdx.x * 256 + threadIdx.x;
  const int a = esrc[e], b = edst[e];
  const float rx = pos[b * 3 + 0] - pos[a * 3 + 0];
  const float ry = pos[b * 3 + 1] - pos[a * 3 + 1];
  const float rz = pos[b * 3 + 2] - pos[a * 3 + 2];
  const float d  = sqrtf(rx * rx + ry * ry + rz * rz + 1e-8f);
  const float id = __builtin_amdgcn_rcpf(d);
  unitv[(size_t)e * 3 + 0] = rx * id;
  unitv[(size_t)e * 3 + 1] = ry * id;
  unitv[(size_t)e * 3 + 2] = rz * id;
  fcv[e] = (d < CUTOFF) ? 0.5f * (__cosf(PI_F * d / CUTOFF) + 1.0f) : 0.0f;
  const float base = PI_F / CUTOFF;
#pragma unroll
  for (int t = 0; t < RDIM; ++t)
    rbfv[(size_t)e * RDIM + t] = __sinf(d * base * (float)(t + 1)) * id;
}

// ---------------------------------------------------------------------------
// Fused edge kernel: wfilt = (rbf @ rbf_w + rbf_b)*fc computed on the fly with
// rbf_w staged in LDS (20x384 = 30KB << 320KB/WGP). One wave = one edge; each
// lane owns 4 features. ds accumulated straight into s (edge pass reads only
// phi); dv into a separate buffer so v[edge_src] stays pre-update.
// ---------------------------------------------------------------------------
__global__ void k_edge_msg(const int* __restrict__ esrc, const int* __restrict__ edst,
                           const float* __restrict__ phi, const float* __restrict__ v,
                           const float* __restrict__ unitv, const float* __restrict__ fcv,
                           const float* __restrict__ rbfv,
                           const float* __restrict__ rbf_w, const float* __restrict__ rbf_b,
                           float* __restrict__ s_acc, float* __restrict__ dv_acc) {
  __shared__ float lw[RDIM * F3];
  __shared__ float lb[F3];
  for (int i = threadIdx.x; i < RDIM * F3; i += 256) lw[i] = rbf_w[i];
  for (int i = threadIdx.x; i < F3; i += 256)        lb[i] = rbf_b[i];
  __syncthreads();

  const int wave = threadIdx.x >> 5;
  const int lane = threadIdx.x & 31;
  const int e    = blockIdx.x * 8 + wave;
  const int src  = esrc[e], dst = edst[e];
  const float fce = fcv[e];
  const float ux = unitv[(size_t)e * 3 + 0];
  const float uy = unitv[(size_t)e * 3 + 1];
  const float uz = unitv[(size_t)e * 3 + 2];

  const float* __restrict__ phis = phi + (size_t)src * F3;
  const float* __restrict__ vsrc = v + (size_t)src * 3 * FDIM;
  __builtin_prefetch(phis, 0, 0);            // global_prefetch_b8
  __builtin_prefetch(vsrc, 0, 0);

  float rb[RDIM];
#pragma unroll
  for (int t = 0; t < RDIM; ++t) rb[t] = rbfv[(size_t)e * RDIM + t];

#pragma unroll
  for (int q = 0; q < 4; ++q) {
    const int f = lane + q * 32;
    float w0 = lb[f], w1 = lb[FDIM + f], w2 = lb[2 * FDIM + f];
#pragma unroll
    for (int t = 0; t < RDIM; ++t) {
      const float r = rb[t];
      w0 = fmaf(r, lw[t * F3 + f],            w0);
      w1 = fmaf(r, lw[t * F3 + FDIM + f],     w1);
      w2 = fmaf(r, lw[t * F3 + 2 * FDIM + f], w2);
    }
    w0 *= fce; w1 *= fce; w2 *= fce;
    const float dsv = phis[f] * w0;            // ds_e
    const float dvv = phis[FDIM + f] * w1;     // gate on v[src]
    const float dvs = phis[2 * FDIM + f] * w2; // gate on unit

    atomicAdd(&s_acc[(size_t)dst * FDIM + f], dsv);
    const float vx = vsrc[0 * FDIM + f];
    const float vy = vsrc[1 * FDIM + f];
    const float vz = vsrc[2 * FDIM + f];
    atomicAdd(&dv_acc[((size_t)dst * 3 + 0) * FDIM + f], fmaf(vx, dvv, ux * dvs));
    atomicAdd(&dv_acc[((size_t)dst * 3 + 1) * FDIM + f], fmaf(vy, dvv, uy * dvs));
    atomicAdd(&dv_acc[((size_t)dst * 3 + 2) * FDIM + f], fmaf(vz, dvv, uz * dvs));
  }
}

__global__ void k_zero(float* __restrict__ p, int n) {
  const int i = blockIdx.x * 256 + threadIdx.x;
  if (i < n) p[i] = 0.0f;
}

__global__ void k_vadd(float* __restrict__ v, const float* __restrict__ dv, int n) {
  const int i = blockIdx.x * 256 + threadIdx.x;
  if (i < n) v[i] += dv[i];
}

// x = concat(s, ||vv||_3) : (8192, 256)
__global__ void k_norm_concat(const float* __restrict__ s, const float* __restrict__ vv,
                              float* __restrict__ x) {
  const int i = blockIdx.x, f = threadIdx.x;
  const float a = vv[((size_t)i * 3 + 0) * FDIM + f];
  const float b = vv[((size_t)i * 3 + 1) * FDIM + f];
  const float c = vv[((size_t)i * 3 + 2) * FDIM + f];
  x[(size_t)i * 256 + f]        = s[(size_t)i * FDIM + f];
  x[(size_t)i * 256 + FDIM + f] = sqrtf(a * a + b * b + c * c + 1e-8f);
}

// s += a_ss + a_sv * <uv,vv>_3 ; v += a_vv * uv
__global__ void k_apply(float* __restrict__ s, float* __restrict__ v,
                        const float* __restrict__ uv, const float* __restrict__ vv,
                        const float* __restrict__ a) {
  const int i = blockIdx.x, f = threadIdx.x;
  const float u0 = uv[((size_t)i * 3 + 0) * FDIM + f];
  const float u1 = uv[((size_t)i * 3 + 1) * FDIM + f];
  const float u2 = uv[((size_t)i * 3 + 2) * FDIM + f];
  const float w0 = vv[((size_t)i * 3 + 0) * FDIM + f];
  const float w1 = vv[((size_t)i * 3 + 1) * FDIM + f];
  const float w2 = vv[((size_t)i * 3 + 2) * FDIM + f];
  const float scal = u0 * w0 + u1 * w1 + u2 * w2;
  const float avv = a[(size_t)i * F3 + f];
  const float asv = a[(size_t)i * F3 + FDIM + f];
  const float ass = a[(size_t)i * F3 + 2 * FDIM + f];
  s[(size_t)i * FDIM + f] += ass + asv * scal;
  v[((size_t)i * 3 + 0) * FDIM + f] += avv * u0;
  v[((size_t)i * 3 + 1) * FDIM + f] += avv * u1;
  v[((size_t)i * 3 + 2) * FDIM + f] += avv * u2;
}

// atom_out = h @ out_w2 + b2 ; segment-sum into mol buckets. One wave / atom.
__global__ void k_readout(const float* __restrict__ h, const float* __restrict__ w2,
                          const float* __restrict__ b2, const int* __restrict__ mol,
                          float* __restrict__ out) {
  const int wave = threadIdx.x >> 5, lane = threadIdx.x & 31;
  const int i = blockIdx.x * 8 + wave;
  float p = 0.0f;
#pragma unroll
  for (int q = 0; q < 4; ++q) {
    const int f = lane + q * 32;
    p = fmaf(h[(size_t)i * FDIM + f], w2[f], p);
  }
#pragma unroll
  for (int off = 16; off > 0; off >>= 1) p += __shfl_xor(p, off, 32);
  if (lane == 0) atomicAdd(&out[mol[i]], p + b2[0]);
}

// ---------------------------------------------------------------------------
extern "C" void kernel_launch(void* const* d_in, const int* in_sizes, int n_in,
                              void* d_out, int out_size, void* d_ws, size_t ws_size,
                              hipStream_t stream) {
  const int*   z       = (const int*)  d_in[0];
  const float* pos     = (const float*)d_in[1];
  const int*   esrc    = (const int*)  d_in[2];
  const int*   edst    = (const int*)  d_in[3];
  const int*   mol     = (const int*)  d_in[4];
  const float* embed   = (const float*)d_in[5];
  const float* msg_w1  = (const float*)d_in[6];
  const float* msg_b1  = (const float*)d_in[7];
  const float* msg_w2  = (const float*)d_in[8];
  const float* msg_b2  = (const float*)d_in[9];
  const float* rbf_w   = (const float*)d_in[10];
  const float* rbf_b   = (const float*)d_in[11];
  const float* upd_u   = (const float*)d_in[12];
  const float* upd_v   = (const float*)d_in[13];
  const float* upd_a1  = (const float*)d_in[14];
  const float* upd_a1b = (const float*)d_in[15];
  const float* upd_a2  = (const float*)d_in[16];
  const float* upd_a2b = (const float*)d_in[17];
  const float* out_w1  = (const float*)d_in[18];
  const float* out_b1  = (const float*)d_in[19];
  const float* out_w2  = (const float*)d_in[20];
  const float* out_b2  = (const float*)d_in[21];
  float* out = (float*)d_out;

  // ---- workspace carve-up (fp32), ~105 MB total ----
  float* w = (float*)d_ws;
  float* s     = w; w += (size_t)N_ATOMS * FDIM;
  float* v     = w; w += (size_t)N_ATOMS * 3 * FDIM;
  float* dv    = w; w += (size_t)N_ATOMS * 3 * FDIM;
  float* hbuf  = w; w += (size_t)N_ATOMS * FDIM;        // MLP hidden (reused)
  float* phi   = w; w += (size_t)N_ATOMS * F3;          // phi, later reused as 'a'
  float* uv    = w; w += (size_t)N_ATOMS * 3 * FDIM;
  float* vvb   = w; w += (size_t)N_ATOMS * 3 * FDIM;
  float* xcat  = w; w += (size_t)N_ATOMS * 256;
  float* rbfv  = w; w += (size_t)N_EDGES * RDIM;
  float* unitv = w; w += (size_t)N_EDGES * 3;
  float* fcv   = w; w += (size_t)N_EDGES;

  const int dv_n = N_ATOMS * 3 * FDIM;

  // init: s = embed[z], v = 0 ; edge geometry (reused all layers)
  k_embed<<<N_ATOMS, FDIM, 0, stream>>>(z, embed, s, v);
  k_edge_geom<<<N_EDGES / 256, 256, 0, stream>>>(pos, esrc, edst, unitv, fcv, rbfv);

  for (int l = 0; l < NLAYER; ++l) {
    // phi = silu(s @ msg_w1 + b1) @ msg_w2 + b2
    k_gemm_wmma<128, true, true><<<dim3(1, N_ATOMS / 64), 128, 0, stream>>>(
        s, msg_w1 + (size_t)l * FDIM * FDIM, msg_b1 + (size_t)l * FDIM, hbuf, FDIM);
    k_gemm_wmma<128, false, true><<<dim3(3, N_ATOMS / 64), 128, 0, stream>>>(
        hbuf, msg_w2 + (size_t)l * FDIM * F3, msg_b2 + (size_t)l * F3, phi, F3);

    // edge pass: fused wfilt + gather + scatter-add
    k_zero<<<(dv_n + 255) / 256, 256, 0, stream>>>(dv, dv_n);
    k_edge_msg<<<N_EDGES / 8, 256, 0, stream>>>(
        esrc, edst, phi, v, unitv, fcv, rbfv,
        rbf_w + (size_t)l * RDIM * F3, rbf_b + (size_t)l * F3, s, dv);
    k_vadd<<<(dv_n + 255) / 256, 256, 0, stream>>>(v, dv, dv_n);

    // update block: uv = v@U, vv = v@V  (v viewed as (24576,128))
    k_gemm_wmma<128, false, false><<<dim3(1, (N_ATOMS * 3) / 64), 128, 0, stream>>>(
        v, upd_u + (size_t)l * FDIM * FDIM, nullptr, uv, FDIM);
    k_gemm_wmma<128, false, false><<<dim3(1, (N_ATOMS * 3) / 64), 128, 0, stream>>>(
        v, upd_v + (size_t)l * FDIM * FDIM, nullptr, vvb, FDIM);
    k_norm_concat<<<N_ATOMS, FDIM, 0, stream>>>(s, vvb, xcat);
    k_gemm_wmma<256, true, true><<<dim3(1, N_ATOMS / 64), 128, 0, stream>>>(
        xcat, upd_a1 + (size_t)l * 256 * FDIM, upd_a1b + (size_t)l * FDIM, hbuf, FDIM);
    k_gemm_wmma<128, false, true><<<dim3(3, N_ATOMS / 64), 128, 0, stream>>>(
        hbuf, upd_a2 + (size_t)l * FDIM * F3, upd_a2b + (size_t)l * F3, phi /*a*/, F3);
    k_apply<<<N_ATOMS, FDIM, 0, stream>>>(s, v, uv, vvb, phi);
  }

  // readout: h = silu(s @ out_w1 + b1); out[mol] += h @ out_w2 + b2
  k_gemm_wmma<128, true, true><<<dim3(1, N_ATOMS / 64), 128, 0, stream>>>(
      s, out_w1, out_b1, hbuf, FDIM);
  k_zero<<<1, 256, 0, stream>>>(out, N_MOL);
  k_readout<<<N_ATOMS / 8, 256, 0, stream>>>(hbuf, out_w2, out_b2, mol, out);
}
